// MultiHeadSelfAttention_3633542332733
// MI455X (gfx1250) — matrix operand order repaired and checked by reading, not compiled, not run on hardware
//
#include <hip/hip_runtime.h>
#include <hip/hip_bf16.h>

// ---------------------------------------------------------------------------
// MI455X (gfx1250) multi-head self-attention, wave32 + V_WMMA_F32_16X16X32_F16
// B=4, T=2048, C=1024, H=16, dh=64. Compute-bound -> f16 WMMA w/ f32 accum.
// Pure-f16 LDS staging copies use GLOBAL_LOAD_ASYNC_TO_LDS_B128 (ASYNCcnt).
// ---------------------------------------------------------------------------

typedef __attribute__((ext_vector_type(16))) _Float16 v16h;
typedef __attribute__((ext_vector_type(8)))  _Float16 v8h;
typedef __attribute__((ext_vector_type(8)))  float    v8f;

union FragH { v16h f; v8h h[2]; };

__device__ __forceinline__ v8f wmma_f16(const v16h& a, const v16h& b, const v8f& c) {
  return __builtin_amdgcn_wmma_f32_16x16x32_f16(false, a, false, b, (short)0, c, false, false);
}

// Async copy 16 bytes global -> LDS (gfx1250, tracked by ASYNCcnt).
__device__ __forceinline__ void async_copy_b128(const void* gptr, void* lds_ptr) {
  const unsigned lds_addr = (unsigned)(unsigned long long)lds_ptr; // low 32 bits = LDS offset
  asm volatile("global_load_async_to_lds_b128 %0, %1, off"
               :: "v"(lds_addr), "v"(gptr) : "memory");
}
__device__ __forceinline__ void wait_asynccnt0() {
  asm volatile("s_wait_asynccnt 0" ::: "memory");
}

constexpr int DMODEL = 1024;
constexpr int NB     = 4;
constexpr int SEQ    = 2048;
constexpr int NHEAD  = 16;
constexpr int DHEAD  = 64;
constexpr int MTOK   = NB * SEQ;   // 8192

// ---------------------------------------------------------------------------
// Kernel 1: fused QKV projection.  Y = x @ W_z + b_z  (z = blockIdx.z)
// Block tile 64(M) x 128(N), K-step 32. 8 waves, each 32x32 (2x2 WMMA tiles).
// Staging converts f32 -> f16, so it stays on the VGPR path.
// Output f16 in [B,H,T,dh] layout.
// ---------------------------------------------------------------------------
__global__ __launch_bounds__(256) void qkv_gemm(
    const float* __restrict__ x,
    const float* __restrict__ wq, const float* __restrict__ wk, const float* __restrict__ wv,
    const float* __restrict__ bq, const float* __restrict__ bk, const float* __restrict__ bv,
    _Float16* __restrict__ qo, _Float16* __restrict__ ko, _Float16* __restrict__ vo)
{
  __shared__ __align__(16) _Float16 As[64][40];    // [m][k], padded row stride
  __shared__ __align__(16) _Float16 Bs[128][40];   // [n][k] (W tile transposed)

  const int z = blockIdx.z;
  const float* W    = (z == 0) ? wq : (z == 1) ? wk : wv;
  const float* bias = (z == 0) ? bq : (z == 1) ? bk : bv;
  _Float16*    Y    = (z == 0) ? qo : (z == 1) ? ko : vo;

  const int nBase = blockIdx.x * 128;
  const int mBase = blockIdx.y * 64;

  const int tid  = threadIdx.x;
  const int wave = tid >> 5;
  const int lane = tid & 31;
  const int lh   = lane >> 4;    // lane half (0/1)
  const int col  = lane & 15;
  const int wm   = (wave >> 2) * 32;   // 0 or 32
  const int wn   = (wave & 3) * 32;    // 0..96

  // C layout: every VGPR row shares column n = base + col  -> bias broadcast
  const float bn0 = bias[nBase + wn + col];
  const float bn1 = bias[nBase + wn + 16 + col];
  v8f acc[2][2];
  for (int i = 0; i < 8; ++i) {
    acc[0][0][i] = bn0; acc[1][0][i] = bn0;
    acc[0][1][i] = bn1; acc[1][1][i] = bn1;
  }

  for (int k0 = 0; k0 < DMODEL; k0 += 32) {
    // Stage A (64x32 f32 -> f16), coalesced float4 loads
    for (int pass = 0; pass < 2; ++pass) {
      const int r = pass * 32 + (tid >> 3);
      const int c = (tid & 7) * 4;
      const float4 v = *(const float4*)(x + (size_t)(mBase + r) * DMODEL + k0 + c);
      As[r][c + 0] = (_Float16)v.x; As[r][c + 1] = (_Float16)v.y;
      As[r][c + 2] = (_Float16)v.z; As[r][c + 3] = (_Float16)v.w;
    }
    // Stage B transposed (W rows k0..k0+31, cols nBase..+127 -> Bs[n][k])
    for (int pass = 0; pass < 4; ++pass) {
      const int kr = pass * 8 + (tid >> 5);
      const int nc = (tid & 31) * 4;
      const float4 v = *(const float4*)(W + (size_t)(k0 + kr) * DMODEL + nBase + nc);
      Bs[nc + 0][kr] = (_Float16)v.x; Bs[nc + 1][kr] = (_Float16)v.y;
      Bs[nc + 2][kr] = (_Float16)v.z; Bs[nc + 3][kr] = (_Float16)v.w;
    }
    __syncthreads();

    FragH a[2], b[2];
    for (int mt = 0; mt < 2; ++mt) {
      const _Float16* ap = &As[wm + mt * 16 + col][0];
      a[mt].h[0] = *(const v8h*)(ap + lh * 8);        // K {0..7}/{8..15}
      a[mt].h[1] = *(const v8h*)(ap + 16 + lh * 8);   // K {16..23}/{24..31}
    }
    for (int nt = 0; nt < 2; ++nt) {
      const _Float16* bp = &Bs[wn + nt * 16 + col][0];
      b[nt].h[0] = *(const v8h*)(bp + lh * 8);        // B: K {0..7}/{8..15}
      b[nt].h[1] = *(const v8h*)(bp + 16 + lh * 8);
    }
    for (int mt = 0; mt < 2; ++mt)
      for (int nt = 0; nt < 2; ++nt)
        acc[mt][nt] = wmma_f16(a[mt].f, b[nt].f, acc[mt][nt]);
    __syncthreads();
  }

  // Store f16 into [B,H,T,dh]
  for (int mt = 0; mt < 2; ++mt)
    for (int nt = 0; nt < 2; ++nt)
      for (int r = 0; r < 8; ++r) {
        const int m = mBase + wm + mt * 16 + r + lh * 8;
        const int n = nBase + wn + nt * 16 + col;
        const int bidx = m >> 11, t = m & (SEQ - 1);
        const int h = n >> 6, d = n & 63;
        Y[(((size_t)bidx * NHEAD + h) * SEQ + t) * DHEAD + d] = (_Float16)acc[mt][nt][r];
      }
}

// ---------------------------------------------------------------------------
// Kernel 2: causal flash attention.  grid = (T/128, B*H)
// 8 waves x 16 query rows; 64-key blocks; online softmax; WMMA for QK^T & PV.
// K tile staged via async global->LDS copy; V tile transposed on the VGPR path.
// ---------------------------------------------------------------------------
__global__ __launch_bounds__(256) void attn_kernel(
    const _Float16* __restrict__ Q, const _Float16* __restrict__ K,
    const _Float16* __restrict__ V, _Float16* __restrict__ O)
{
  __shared__ __align__(16) _Float16 Ks[64][72];      // [tk][d]
  __shared__ __align__(16) _Float16 Vts[64][72];     // [d][tk] (transposed)
  __shared__ __align__(16) _Float16 Ps[8][16][64];   // per-wave P staging

  const int bh    = blockIdx.y;
  const int qBase = blockIdx.x * 128;
  const int tid   = threadIdx.x;
  const int wave  = tid >> 5;
  const int lane  = tid & 31;
  const int lh    = lane >> 4;
  const int col   = lane & 15;

  const _Float16* Qb = Q + (size_t)bh * SEQ * DHEAD;
  const _Float16* Kb = K + (size_t)bh * SEQ * DHEAD;
  const _Float16* Vb = V + (size_t)bh * SEQ * DHEAD;

  // Q fragments for this wave's 16 rows: loaded once, stay in registers
  FragH qf[2];
  {
    const _Float16* qp = Qb + (size_t)(qBase + wave * 16 + col) * DHEAD;
    for (int f = 0; f < 2; ++f) {
      qf[f].h[0] = *(const v8h*)(qp + f * 32 + lh * 8);
      qf[f].h[1] = *(const v8h*)(qp + f * 32 + 16 + lh * 8);
    }
  }

  v8f o[4];
  for (int dt = 0; dt < 4; ++dt) for (int i = 0; i < 8; ++i) o[dt][i] = 0.f;
  float mrow[8], lrow[8];
  for (int r = 0; r < 8; ++r) { mrow[r] = -1e30f; lrow[r] = 0.f; }

  const int nkb = blockIdx.x * 2 + 2;   // causal: only key blocks <= query tile
  for (int kb = 0; kb < nkb; ++kb) {
    const int k0 = kb * 64;
    // Cooperative stage: K row-major via async copy (no VGPR round-trip),
    // V transposed on the VGPR path (needs the shuffle anyway).
    for (int pass = 0; pass < 2; ++pass) {
      const int idx = pass * 256 + tid;
      const int row = idx >> 3;
      const int c8  = (idx & 7) * 8;
      async_copy_b128(Kb + (size_t)(k0 + row) * DHEAD + c8, &Ks[row][c8]);
      const v8h vv = *(const v8h*)(Vb + (size_t)(k0 + row) * DHEAD + c8);
      for (int j = 0; j < 8; ++j) Vts[c8 + j][row] = vv[j];
    }
    wait_asynccnt0();
    __syncthreads();

    // S = Q @ K^T  (4 key sub-tiles of 16)
    v8f s[4];
    for (int nt = 0; nt < 4; ++nt) {
      v8f zacc = {0.f, 0.f, 0.f, 0.f, 0.f, 0.f, 0.f, 0.f};
      const _Float16* kp = &Ks[nt * 16 + col][0];
      FragH bf0, bf1;
      bf0.h[0] = *(const v8h*)(kp + lh * 8);
      bf0.h[1] = *(const v8h*)(kp + 16 + lh * 8);
      bf1.h[0] = *(const v8h*)(kp + 32 + lh * 8);
      bf1.h[1] = *(const v8h*)(kp + 32 + 16 + lh * 8);
      zacc = wmma_f16(qf[0].f, bf0.f, zacc);
      zacc = wmma_f16(qf[1].f, bf1.f, zacc);
      s[nt] = zacc;
    }

    // scale + causal mask
    for (int nt = 0; nt < 4; ++nt)
      for (int r = 0; r < 8; ++r) {
        const int tq = qBase + wave * 16 + r + lh * 8;
        const int tk = k0 + nt * 16 + col;
        const float v = s[nt][r] * 0.125f;            // 1/sqrt(64)
        s[nt][r] = (tk <= tq) ? v : -1e30f;
      }

    // online softmax: row reductions across the 16-lane half holding each row
    for (int r = 0; r < 8; ++r) {
      float mx = fmaxf(fmaxf(s[0][r], s[1][r]), fmaxf(s[2][r], s[3][r]));
      for (int off = 1; off < 16; off <<= 1) mx = fmaxf(mx, __shfl_xor(mx, off, 32));
      const float mnew  = fmaxf(mrow[r], mx);
      const float alpha = __expf(mrow[r] - mnew);
      mrow[r] = mnew;
      float rs = 0.f;
      for (int nt = 0; nt < 4; ++nt) {
        const float p = __expf(s[nt][r] - mnew);
        s[nt][r] = p;
        rs += p;
      }
      for (int off = 1; off < 16; off <<= 1) rs += __shfl_xor(rs, off, 32);
      lrow[r] = lrow[r] * alpha + rs;
      for (int dt = 0; dt < 4; ++dt) o[dt][r] *= alpha;
    }

    // C-layout -> A-layout via per-wave LDS patch
    for (int nt = 0; nt < 4; ++nt)
      for (int r = 0; r < 8; ++r)
        Ps[wave][r + lh * 8][nt * 16 + col] = (_Float16)s[nt][r];

    FragH pf[2];
    {
      const _Float16* pp = &Ps[wave][col][0];
      for (int f = 0; f < 2; ++f) {
        pf[f].h[0] = *(const v8h*)(pp + f * 32 + lh * 8);
        pf[f].h[1] = *(const v8h*)(pp + f * 32 + 16 + lh * 8);
      }
    }

    // O += P @ V
    for (int dt = 0; dt < 4; ++dt) {
      const _Float16* vp = &Vts[dt * 16 + col][0];
      FragH vf0, vf1;
      vf0.h[0] = *(const v8h*)(vp + lh * 8);
      vf0.h[1] = *(const v8h*)(vp + 16 + lh * 8);
      vf1.h[0] = *(const v8h*)(vp + 32 + lh * 8);
      vf1.h[1] = *(const v8h*)(vp + 32 + 16 + lh * 8);
      o[dt] = wmma_f16(pf[0].f, vf0.f, o[dt]);
      o[dt] = wmma_f16(pf[1].f, vf1.f, o[dt]);
    }
    __syncthreads();
  }

  // epilogue: normalize, write f16 [B,T,C] for the output projection
  const int b = bh >> 4, h = bh & 15;
  for (int r = 0; r < 8; ++r) {
    const float inv = 1.f / lrow[r];
    const int tq = qBase + wave * 16 + r + lh * 8;
    _Float16* op = O + ((size_t)b * SEQ + tq) * DMODEL + h * DHEAD;
    for (int dt = 0; dt < 4; ++dt)
      op[dt * 16 + col] = (_Float16)(o[dt][r] * inv);
  }
}

// ---------------------------------------------------------------------------
// Kernel 3: output projection.  out = attn(f16) @ wo + bo  (f32 result)
// A tile is a pure f16 copy -> async global->LDS path.
// ---------------------------------------------------------------------------
__global__ __launch_bounds__(256) void out_gemm(
    const _Float16* __restrict__ A, const float* __restrict__ W,
    const float* __restrict__ bias, float* __restrict__ Y)
{
  __shared__ __align__(16) _Float16 As[64][40];
  __shared__ __align__(16) _Float16 Bs[128][40];

  const int nBase = blockIdx.x * 128;
  const int mBase = blockIdx.y * 64;
  const int tid  = threadIdx.x;
  const int wave = tid >> 5;
  const int lane = tid & 31;
  const int lh   = lane >> 4;
  const int col  = lane & 15;
  const int wm   = (wave >> 2) * 32;
  const int wn   = (wave & 3) * 32;

  const float bn0 = bias[nBase + wn + col];
  const float bn1 = bias[nBase + wn + 16 + col];
  v8f acc[2][2];
  for (int i = 0; i < 8; ++i) {
    acc[0][0][i] = bn0; acc[1][0][i] = bn0;
    acc[0][1][i] = bn1; acc[1][1][i] = bn1;
  }

  for (int k0 = 0; k0 < DMODEL; k0 += 32) {
    { // A already f16: 64 rows x 32 halfs, async b128 copies
      const int r = tid >> 2;
      const int c = (tid & 3) * 8;
      async_copy_b128(A + (size_t)(mBase + r) * DMODEL + k0 + c, &As[r][c]);
    }
    for (int pass = 0; pass < 4; ++pass) {
      const int kr = pass * 8 + (tid >> 5);
      const int nc = (tid & 31) * 4;
      const float4 v = *(const float4*)(W + (size_t)(k0 + kr) * DMODEL + nBase + nc);
      Bs[nc + 0][kr] = (_Float16)v.x; Bs[nc + 1][kr] = (_Float16)v.y;
      Bs[nc + 2][kr] = (_Float16)v.z; Bs[nc + 3][kr] = (_Float16)v.w;
    }
    wait_asynccnt0();
    __syncthreads();

    FragH a[2], b[2];
    for (int mt = 0; mt < 2; ++mt) {
      const _Float16* ap = &As[wm + mt * 16 + col][0];
      a[mt].h[0] = *(const v8h*)(ap + lh * 8);
      a[mt].h[1] = *(const v8h*)(ap + 16 + lh * 8);
    }
    for (int nt = 0; nt < 2; ++nt) {
      const _Float16* bp = &Bs[wn + nt * 16 + col][0];
      b[nt].h[0] = *(const v8h*)(bp + lh * 8);
      b[nt].h[1] = *(const v8h*)(bp + 16 + lh * 8);
    }
    for (int mt = 0; mt < 2; ++mt)
      for (int nt = 0; nt < 2; ++nt)
        acc[mt][nt] = wmma_f16(a[mt].f, b[nt].f, acc[mt][nt]);
    __syncthreads();
  }

  for (int mt = 0; mt < 2; ++mt)
    for (int nt = 0; nt < 2; ++nt)
      for (int r = 0; r < 8; ++r) {
        const int m = mBase + wm + mt * 16 + r + lh * 8;
        const int n = nBase + wn + nt * 16 + col;
        Y[(size_t)m * DMODEL + n] = acc[mt][nt][r];
      }
}

// ---------------------------------------------------------------------------
extern "C" void kernel_launch(void* const* d_in, const int* in_sizes, int n_in,
                              void* d_out, int out_size, void* d_ws, size_t ws_size,
                              hipStream_t stream)
{
  (void)in_sizes; (void)n_in; (void)out_size; (void)ws_size;
  const float* x  = (const float*)d_in[0];
  // d_in[1] = causal mask (bool) -> causality is applied analytically
  const float* wq = (const float*)d_in[2];
  const float* bq = (const float*)d_in[3];
  const float* wk = (const float*)d_in[4];
  const float* bk = (const float*)d_in[5];
  const float* wv = (const float*)d_in[6];
  const float* bv = (const float*)d_in[7];
  const float* wo = (const float*)d_in[8];
  const float* bo = (const float*)d_in[9];

  // Workspace: Q,K,V f16 [B,H,T,dh] (16 MB each) + attn f16 [B,T,C] (16 MB)
  char* ws = (char*)d_ws;
  _Float16* Qh = (_Float16*)(ws + (size_t)0);
  _Float16* Kh = (_Float16*)(ws + ((size_t)16 << 20));
  _Float16* Vh = (_Float16*)(ws + ((size_t)32 << 20));
  _Float16* Ah = (_Float16*)(ws + ((size_t)48 << 20));

  qkv_gemm<<<dim3(DMODEL / 128, MTOK / 64, 3), 256, 0, stream>>>(
      x, wq, wk, wv, bq, bk, bv, Qh, Kh, Vh);
  attn_kernel<<<dim3(SEQ / 128, NB * NHEAD, 1), 256, 0, stream>>>(Qh, Kh, Vh, Ah);
  out_gemm<<<dim3(DMODEL / 128, MTOK / 64, 1), 256, 0, stream>>>(
      Ah, wo, bo, (float*)d_out);
}
